// SparseLinearAttention_72146860638377
// MI455X (gfx1250) — matrix operand order repaired and checked by reading, not compiled, not run on hardware
//
#include <hip/hip_runtime.h>

typedef _Float16 f16;
typedef __attribute__((ext_vector_type(16))) _Float16 v16h;
typedef __attribute__((ext_vector_type(8)))  _Float16 v8h;
typedef __attribute__((ext_vector_type(2)))  _Float16 v2h;
typedef __attribute__((ext_vector_type(8)))  float    v8f;

#define HN   16
#define LN   8192
#define DN   64
#define BQ   64
#define BKV  64
#define MB   (LN / BQ)    // 128
#define NB   (LN / BKV)   // 128
#define TOPK 16
#define LDST 72           // padded LDS row stride (halves); 144 B = 9*16 keeps 16B alignment

// ---------------- Phase 1: block sums for q and k ----------------
__global__ void pool_sums(const float* __restrict__ q, const float* __restrict__ k,
                          float* __restrict__ qsum, float* __restrict__ ksum) {
    const int mb = blockIdx.x, h = blockIdx.y, d = threadIdx.x; // 64 threads
    const float* qb = q + ((size_t)h * LN + (size_t)mb * BQ) * DN + d;
    const float* kb = k + ((size_t)h * LN + (size_t)mb * BKV) * DN + d;
    float sq = 0.f, sk = 0.f;
    for (int r = 0; r < BQ; ++r) { sq += qb[r * DN]; sk += kb[r * DN]; }
    qsum[((size_t)h * MB + mb) * DN + d] = sq;
    ksum[((size_t)h * NB + mb) * DN + d] = sk;
}

// ------- Phase 1b: q_pool = qsum/64 ; k_pool = ksum/64 - mean(k) -------
__global__ void pool_finish(float* __restrict__ qsum, float* __restrict__ ksum) {
    const int h = blockIdx.x, d = threadIdx.x; // 64 threads
    float* kr = ksum + (size_t)h * NB * DN + d;
    float* qr = qsum + (size_t)h * MB * DN + d;
    float tot = 0.f;
    for (int n = 0; n < NB; ++n) tot += kr[n * DN];
    const float meank = tot / (float)LN;
    for (int n = 0; n < NB; ++n) {
        kr[n * DN] = kr[n * DN] * (1.0f / BKV) - meank;
        qr[n * DN] = qr[n * DN] * (1.0f / BQ);
    }
}

// ---------------- Phase 2: block scores + top-16 LUT ----------------
__global__ void block_topk(const float* __restrict__ qpool, const float* __restrict__ kpool,
                           int* __restrict__ lut) {
    const int mb = blockIdx.x, h = blockIdx.y, tid = threadIdx.x; // 128 threads
    __shared__ float qv[DN];
    __shared__ float score[NB];
    __shared__ float rv[64];
    __shared__ int   ri[64];
    if (tid < DN) qv[tid] = qpool[((size_t)h * MB + mb) * DN + tid];
    __syncthreads();
    const float* kr = kpool + ((size_t)h * NB + tid) * DN;
    float s = 0.f;
    for (int d = 0; d < DN; ++d) s += qv[d] * kr[d];
    score[tid] = s;
    __syncthreads();
    for (int t = 0; t < TOPK; ++t) {
        if (tid < 64) {
            float a = score[tid], b = score[tid + 64];
            bool take = b > a;                       // tie -> lower index (matches jax)
            rv[tid] = take ? b : a;
            ri[tid] = take ? tid + 64 : tid;
        }
        __syncthreads();
        for (int s2 = 32; s2 > 0; s2 >>= 1) {
            if (tid < s2) {
                if (rv[tid + s2] > rv[tid]) { rv[tid] = rv[tid + s2]; ri[tid] = ri[tid + s2]; }
            }
            __syncthreads();
        }
        if (tid == 0) {
            int w = ri[0];
            lut[((size_t)h * MB + mb) * TOPK + t] = w;
            score[w] = -3.0e38f;
        }
        __syncthreads();
    }
}

// ---------------- Phase 3: sparse flash-attention with WMMA ----------------
__launch_bounds__(128)
__global__ void sparse_attn(const float* __restrict__ q, const float* __restrict__ k,
                            const float* __restrict__ v, const int* __restrict__ lut,
                            float* __restrict__ out) {
    const int mb = blockIdx.x, h = blockIdx.y;
    const int tid  = threadIdx.x;
    const int wave = tid >> 5, lane = tid & 31;
    const int hgrp = (lane >> 4) & 1;   // half-group: lanes 0-15 vs 16-31
    const int lm   = lane & 15;
    const int abase = 8 * hgrp;          // A-frag K-offset base (paired-segment layout)
    const int bbase = 8 * hgrp;          // B-frag K-offset base (paired-segment layout)
    const int mslab = wave * 16;         // this wave's 16-row query slab

    __shared__ f16 k_lds[BKV * LDST];    // K block, row-major  [key][d]
    __shared__ f16 vt_lds[DN * LDST];    // V block, transposed [d][key]
    __shared__ f16 p_lds[BQ * LDST];     // P staging           [m][key]

    union Frag { v16h v; f16 h[16]; };

    // staging sub-tile: each thread owns rows {r0, r0+1} x cols [c0, c0+16)
    const int r0 = (tid >> 2) * 2;       // 0..62 even
    const int c0 = (tid & 3) * 16;       // 0,16,32,48

    // ---- Q slab -> A fragments (2 chunks of K=32), pre-scaled by D^-0.5 ----
    const float sm_scale = 0.125f;
    Frag aq[2];
    {
        const int row = mb * BQ + mslab + lm;
        const float* qr = q + ((size_t)h * LN + row) * DN;
        #pragma unroll
        for (int c = 0; c < 2; ++c) {
            #pragma unroll
            for (int j = 0; j < 8; ++j) aq[c].h[j]     = (f16)(qr[c * 32 + abase + j]      * sm_scale);
            #pragma unroll
            for (int j = 0; j < 8; ++j) aq[c].h[8 + j] = (f16)(qr[c * 32 + abase + 16 + j] * sm_scale);
        }
    }

    v8f o_acc[4];
    #pragma unroll
    for (int i = 0; i < 4; ++i) o_acc[i] = (v8f){};
    float row_m[8], row_l[8];
    #pragma unroll
    for (int r = 0; r < 8; ++r) { row_m[r] = -3.0e38f; row_l[r] = 0.f; }

    const int* lrow = lut + ((size_t)h * MB + mb) * TOPK;

    for (int t = 0; t < TOPK; ++t) {
        const int kv = lrow[t];
        const float* kbase = k + ((size_t)h * LN + (size_t)kv * BKV) * DN;
        const float* vbase = v + ((size_t)h * LN + (size_t)kv * BKV) * DN;

        // ---- cooperative stage: K (row-major, b128 stores) and V^T (packed b32) ----
        __syncthreads();                  // prior iteration done reading LDS
        {
            #pragma unroll
            for (int rr = 0; rr < 2; ++rr) {
                const float* src = kbase + (size_t)(r0 + rr) * DN + c0;
                v8h h0, h1;
                #pragma unroll
                for (int j = 0; j < 8; ++j) { h0[j] = (f16)src[j]; h1[j] = (f16)src[8 + j]; }
                *(v8h*)&k_lds[(r0 + rr) * LDST + c0]     = h0;
                *(v8h*)&k_lds[(r0 + rr) * LDST + c0 + 8] = h1;
            }
            const float* s0 = vbase + (size_t)r0 * DN + c0;
            const float* s1 = vbase + (size_t)(r0 + 1) * DN + c0;
            #pragma unroll
            for (int j = 0; j < 16; ++j) {
                v2h p; p[0] = (f16)s0[j]; p[1] = (f16)s1[j];
                *(v2h*)&vt_lds[(c0 + j) * LDST + r0] = p;   // [d][key], key pair packed
            }
        }
        __syncthreads();

        // ---- prefetch next gathered KV block into cache (global_prefetch_b8) ----
        if (t + 1 < TOPK) {
            const int kn = lrow[t + 1];
            const float* kp = k + ((size_t)h * LN + (size_t)kn * BKV + (tid >> 1)) * DN + (tid & 1) * 32;
            const float* vp = v + ((size_t)h * LN + (size_t)kn * BKV + (tid >> 1)) * DN + (tid & 1) * 32;
            __builtin_prefetch(kp, 0, 0);
            __builtin_prefetch(vp, 0, 0);
        }

        // ---- S = (Q*scale) . K^T : 4 N-tiles x 2 K-chunks ----
        v8f s_acc[4];
        #pragma unroll
        for (int nt = 0; nt < 4; ++nt) {
            v8f acc = (v8f){};
            #pragma unroll
            for (int c = 0; c < 2; ++c) {
                Frag bk;
                const f16* src = &k_lds[(nt * 16 + lm) * LDST + c * 32 + bbase];
                #pragma unroll
                for (int j = 0; j < 16; ++j) bk.h[j] = src[j + (j & 8)];
                acc = __builtin_amdgcn_wmma_f32_16x16x32_f16(false, aq[c].v, false, bk.v,
                                                             (short)0, acc, false, false);
            }
            s_acc[nt] = acc;
        }

        // ---- online softmax: rows are (reg r, half-group) ----
        float mnew[8], alpha[8];
        #pragma unroll
        for (int r = 0; r < 8; ++r) {
            float m = fmaxf(fmaxf(s_acc[0][r], s_acc[1][r]), fmaxf(s_acc[2][r], s_acc[3][r]));
            #pragma unroll
            for (int off = 1; off < 16; off <<= 1)
                m = fmaxf(m, __shfl_xor(m, off, 32));
            mnew[r]  = fmaxf(row_m[r], m);
            alpha[r] = __expf(row_m[r] - mnew[r]);
            row_m[r] = mnew[r];
        }
        #pragma unroll
        for (int r = 0; r < 8; ++r) {
            float rs = 0.f;
            #pragma unroll
            for (int nt = 0; nt < 4; ++nt) {
                float pv = __expf(s_acc[nt][r] - mnew[r]);
                rs += pv;
                p_lds[(mslab + r + 8 * hgrp) * LDST + nt * 16 + lm] = (f16)pv;
            }
            #pragma unroll
            for (int off = 1; off < 16; off <<= 1)
                rs += __shfl_xor(rs, off, 32);
            row_l[r] = row_l[r] * alpha[r] + rs;
            #pragma unroll
            for (int dt = 0; dt < 4; ++dt) o_acc[dt][r] *= alpha[r];
        }

        // ---- P D-layout -> A-layout via LDS (wave-private rows, DS in-order) ----
        Frag ap[2];
        {
            const f16* pr = &p_lds[(mslab + lm) * LDST];
            #pragma unroll
            for (int c = 0; c < 2; ++c) {
                #pragma unroll
                for (int j = 0; j < 8; ++j) ap[c].h[j]     = pr[c * 32 + abase + j];
                #pragma unroll
                for (int j = 0; j < 8; ++j) ap[c].h[8 + j] = pr[c * 32 + abase + 16 + j];
            }
        }

        // ---- O += P . V : 4 D-tiles x 2 K-chunks (B rows come from V^T) ----
        #pragma unroll
        for (int dt = 0; dt < 4; ++dt) {
            v8f acc = o_acc[dt];
            #pragma unroll
            for (int c = 0; c < 2; ++c) {
                Frag bv;
                const f16* src = &vt_lds[(dt * 16 + lm) * LDST + c * 32 + bbase];
                #pragma unroll
                for (int j = 0; j < 16; ++j) bv.h[j] = src[j + (j & 8)];
                acc = __builtin_amdgcn_wmma_f32_16x16x32_f16(false, ap[c].v, false, bv.v,
                                                             (short)0, acc, false, false);
            }
            o_acc[dt] = acc;
        }
    }

    // ---- epilogue: O /= l, store f32 ----
    {
        const int row0 = mb * BQ + mslab + 8 * hgrp;
        #pragma unroll
        for (int r = 0; r < 8; ++r) {
            const float inv = 1.0f / row_l[r];
            float* orow = out + ((size_t)h * LN + row0 + r) * DN + lm;
            #pragma unroll
            for (int dt = 0; dt < 4; ++dt) orow[dt * 16] = o_acc[dt][r] * inv;
        }
    }
}

extern "C" void kernel_launch(void* const* d_in, const int* in_sizes, int n_in,
                              void* d_out, int out_size, void* d_ws, size_t ws_size,
                              hipStream_t stream) {
    (void)in_sizes; (void)n_in; (void)out_size; (void)ws_size;
    const float* q = (const float*)d_in[0];
    const float* k = (const float*)d_in[1];
    const float* v = (const float*)d_in[2];
    float* out = (float*)d_out;

    float* qsum = (float*)d_ws;                        // H*MB*D floats
    float* ksum = qsum + (size_t)HN * MB * DN;         // H*NB*D floats
    int*   lut  = (int*)(ksum + (size_t)HN * NB * DN); // H*MB*TOPK ints

    pool_sums  <<<dim3(MB, HN),  64, 0, stream>>>(q, k, qsum, ksum);
    pool_finish<<<HN,            64, 0, stream>>>(qsum, ksum);
    block_topk <<<dim3(MB, HN), 128, 0, stream>>>(qsum, ksum, lut);
    sparse_attn<<<dim3(MB, HN), 128, 0, stream>>>(q, k, v, lut, out);
}
